// BioNER_Model_73821897883835
// MI455X (gfx1250) — matrix operand order repaired and checked by reading, not compiled, not run on hardware
//
#include <hip/hip_runtime.h>
#include <math.h>

typedef _Float16 v16h __attribute__((ext_vector_type(16)));
typedef _Float16 h8   __attribute__((ext_vector_type(8)));
typedef float    v8f  __attribute__((ext_vector_type(8)));

#define B_SZ 512
#define S_SZ 1024
#define T_SZ 48
#define KPAD 64

// ws layout (floats):
//   [0, 64)            : Mrow[t] = max_j trans[t][j]   (48 used, padded)
//   [64, 64+48*64)     : E[t][j] = exp(trans[t][j]-Mrow[t]), zero-padded to KPAD
//   [3136, 3136+512)   : denom[b]
//   [3648, 3648+512)   : numer[b]
#define WS_MROW  0
#define WS_E     64
#define WS_DEN   3136
#define WS_NUM   3648

// ---------------------------------------------------------------------------
// Kernel 0: precompute row-max and exp'd transition matrix (B operand source)
// ---------------------------------------------------------------------------
__global__ __launch_bounds__(64) void crf_prep(const float* __restrict__ trans,
                                               float* __restrict__ ws) {
  int t = threadIdx.x;
  float* Mrow = ws + WS_MROW;
  float* E    = ws + WS_E;
  if (t < T_SZ) {
    float m = -INFINITY;
    for (int j = 0; j < T_SZ; ++j) m = fmaxf(m, trans[t * T_SZ + j]);
    Mrow[t] = m;
    for (int j = 0; j < KPAD; ++j)
      E[t * KPAD + j] = (j < T_SZ) ? __expf(trans[t * T_SZ + j] - m) : 0.0f;
  } else {
    Mrow[t] = 0.0f;  // pad
  }
}

// ---------------------------------------------------------------------------
// Kernel 1: forward algorithm (denominator). 32 blocks x 96 threads (3 waves).
// Wave w computes the 16x16 tile: 16 batches x tags [16w, 16w+16).
// Inner logsumexp is a GEMM in exp-space -> v_wmma_f32_16x16x32_f16 x2 (K=64).
// ---------------------------------------------------------------------------
__global__ __launch_bounds__(96) void crf_forward(
    const float* __restrict__ emis, const float* __restrict__ mask,
    const float* __restrict__ start_t, const float* __restrict__ end_t,
    const float* __restrict__ ws_const, float* __restrict__ denom) {
  __shared__ float score[16][T_SZ];
  __shared__ float smax[16];
  __shared__ float pmax[16][6];
  __shared__ __align__(16) _Float16 pA[16][KPAD];

  const int tid  = threadIdx.x;
  const int wave = tid >> 5;   // wave32: 0..2
  const int lane = tid & 31;
  const int b0   = blockIdx.x * 16;

  const float* Mrow = ws_const + WS_MROW;
  const float* E    = ws_const + WS_E;

  const int n    = lane & 15;  // N (tag) within tile
  const int g    = lane >> 4;  // lane group
  const int tcol = wave * 16 + n;

  // ---- load resident B fragments (E^T), per documented 16-bit B striping:
  //      lane group g, half h -> K = 8g + h (h<8), 16 + 8g + (h-8) (h>=8)
  v16h Blo, Bhi;
  for (int h = 0; h < 16; ++h) {
    Blo[h] = (_Float16)E[tcol * KPAD + (8 * g + (h & 7) + ((h >> 3) << 4))];
    Bhi[h] = (_Float16)E[tcol * KPAD + (32 + 8 * g + (h & 7) + ((h >> 3) << 4))];
  }
  const float Mt = Mrow[tcol];

  // ---- init score[b][t] = start[t] + emissions[b,0,t]
  for (int i = tid; i < 16 * T_SZ; i += 96) {
    int b = i / T_SZ, t = i % T_SZ;
    score[b][t] = start_t[t] + emis[((size_t)(b0 + b) * S_SZ) * T_SZ + t];
  }
  // zero the K-pad columns of the A panel once (stay zero forever)
  for (int i = tid; i < 16 * (KPAD - T_SZ); i += 96) {
    int b = i / (KPAD - T_SZ), k = T_SZ + (i % (KPAD - T_SZ));
    pA[b][k] = (_Float16)0.0f;
  }
  __syncthreads();

  const int brow_base = 8 * g;  // rows this lane's D fragment covers
  const int mrowA = lane & 15;  // A fragment row (batch) for this lane

  for (int s = 1; s < S_SZ; ++s) {
    // prefetch next step's emissions row segment for this wave's tag tile
    if (s + 1 < S_SZ) {
      __builtin_prefetch(
          emis + ((size_t)(b0 + mrowA) * S_SZ + (s + 1)) * T_SZ + wave * 16, 0, 3);
    }

    // 1) per-batch max over 48 tags: 96 threads = 16 batches x 6 chunks of 8
    {
      int b = tid & 15, c = tid >> 4;
      float m = -INFINITY;
      for (int j = c * 8; j < c * 8 + 8; ++j) m = fmaxf(m, score[b][j]);
      pmax[b][c] = m;
    }
    __syncthreads();
    if (tid < 16) {
      float m = pmax[tid][0];
      for (int c = 1; c < 6; ++c) m = fmaxf(m, pmax[tid][c]);
      smax[tid] = m;
    }
    __syncthreads();

    // 2) A panel: p = exp(score - smax) in f16, zero-padded K
    {
      int b = tid & 15, c = tid >> 4;
      float m = smax[b];
      for (int j = c * 8; j < c * 8 + 8; ++j)
        pA[b][j] = (_Float16)__expf(score[b][j] - m);
    }
    __syncthreads();

    // 3) WMMA: documented 16-bit A layout, lane group g:
    //    halves 0..7 -> K = 8g+h ; halves 8..15 -> K = 8g+8+h
    h8 a0 = *(const h8*)&pA[mrowA][8 * g];
    h8 a1 = *(const h8*)&pA[mrowA][8 * g + 16];
    h8 a2 = *(const h8*)&pA[mrowA][32 + 8 * g];
    h8 a3 = *(const h8*)&pA[mrowA][32 + 8 * g + 16];
    v16h Alo = __builtin_shufflevector(a0, a1, 0, 1, 2, 3, 4, 5, 6, 7, 8, 9, 10,
                                       11, 12, 13, 14, 15);
    v16h Ahi = __builtin_shufflevector(a2, a3, 0, 1, 2, 3, 4, 5, 6, 7, 8, 9, 10,
                                       11, 12, 13, 14, 15);
    v8f acc = {};
    acc = __builtin_amdgcn_wmma_f32_16x16x32_f16(false, Ahi, false, Bhi,
                                                 (short)0, acc, false, false);
    acc = __builtin_amdgcn_wmma_f32_16x16x32_f16(false, Alo, false, Blo,
                                                 (short)0, acc, false, false);

    __syncthreads();  // all A/score reads done before score is overwritten

    // 4) score[b][tcol] = smax + Mt + log(acc) + em, gated by mask
    for (int r = 0; r < 8; ++r) {
      int brow = brow_base + r;          // C/D layout: M = vgpr + 8*(lane/16)
      int bg = b0 + brow;
      float msk = mask[(size_t)bg * S_SZ + s];
      float em  = emis[((size_t)bg * S_SZ + s) * T_SZ + tcol];
      float nv  = smax[brow] + Mt + __logf(fmaxf((float)acc[r], 1e-30f)) + em;
      float old = score[brow][tcol];     // unique (brow,tcol) per thread
      score[brow][tcol] = (msk > 0.0f) ? nv : old;
    }
    __syncthreads();
  }

  // denominator[b] = logsumexp_t(score[b][t] + end[t])
  if (tid < 16) {
    int bg = b0 + tid;
    float m = -INFINITY;
    for (int t = 0; t < T_SZ; ++t) m = fmaxf(m, score[tid][t] + end_t[t]);
    float sum = 0.0f;
    for (int t = 0; t < T_SZ; ++t) sum += __expf(score[tid][t] + end_t[t] - m);
    denom[bg] = m + __logf(sum);
  }
}

// ---------------------------------------------------------------------------
// Kernel 2: numerator (gold-path score). One block per batch.
// ---------------------------------------------------------------------------
__global__ __launch_bounds__(256) void crf_numer(
    const float* __restrict__ emis, const int* __restrict__ tags,
    const float* __restrict__ mask, const float* __restrict__ trans,
    const float* __restrict__ start_t, const float* __restrict__ end_t,
    float* __restrict__ numer) {
  __shared__ float red[256];
  __shared__ float redm[256];
  const int b = blockIdx.x, tid = threadIdx.x;
  const int* tg = tags + (size_t)b * S_SZ;
  float acc = 0.0f, mlen = 0.0f;
  for (int s = tid; s < S_SZ; s += 256) {
    float m = mask[(size_t)b * S_SZ + s];
    mlen += m;
    if (s >= 1) {
      int tc = tg[s], tp = tg[s - 1];
      float em = emis[((size_t)b * S_SZ + s) * T_SZ + tc];
      acc += m * (em + trans[tc * T_SZ + tp]);
    }
  }
  red[tid] = acc;
  redm[tid] = mlen;
  __syncthreads();
  for (int st = 128; st > 0; st >>= 1) {
    if (tid < st) {
      red[tid] += red[tid + st];
      redm[tid] += redm[tid + st];
    }
    __syncthreads();
  }
  if (tid == 0) {
    int len = (int)(redm[0] + 0.5f);
    int last = len - 1;
    if (last < 0) last = 0;
    int t0 = tg[0], tl = tg[last];
    numer[b] = red[0] + start_t[t0] + emis[((size_t)b * S_SZ) * T_SZ + t0] +
               end_t[tl];
  }
}

// ---------------------------------------------------------------------------
// Kernel 3: deterministic final mean reduction
// ---------------------------------------------------------------------------
__global__ __launch_bounds__(512) void crf_final(const float* __restrict__ denom,
                                                 const float* __restrict__ numer,
                                                 float* __restrict__ out) {
  __shared__ float red[512];
  const int tid = threadIdx.x;
  red[tid] = denom[tid] - numer[tid];
  __syncthreads();
  for (int st = 256; st > 0; st >>= 1) {
    if (tid < st) red[tid] += red[tid + st];
    __syncthreads();
  }
  if (tid == 0) out[0] = red[0] / (float)B_SZ;
}

// ---------------------------------------------------------------------------
extern "C" void kernel_launch(void* const* d_in, const int* in_sizes, int n_in,
                              void* d_out, int out_size, void* d_ws,
                              size_t ws_size, hipStream_t stream) {
  (void)in_sizes; (void)n_in; (void)out_size; (void)ws_size;
  const float* emis    = (const float*)d_in[0];  // (B,S,T) f32
  const int*   tags    = (const int*)d_in[1];    // (B,S)   i32
  const float* mask    = (const float*)d_in[2];  // (B,S)   f32
  const float* trans   = (const float*)d_in[3];  // (T,T)   f32
  const float* start_t = (const float*)d_in[4];  // (T,)    f32
  const float* end_t   = (const float*)d_in[5];  // (T,)    f32
  float* ws  = (float*)d_ws;
  float* out = (float*)d_out;

  crf_prep<<<1, 64, 0, stream>>>(trans, ws);
  crf_forward<<<B_SZ / 16, 96, 0, stream>>>(emis, mask, start_t, end_t, ws,
                                            ws + WS_DEN);
  crf_numer<<<B_SZ, 256, 0, stream>>>(emis, tags, mask, trans, start_t, end_t,
                                      ws + WS_NUM);
  crf_final<<<1, 512, 0, stream>>>(ws + WS_DEN, ws + WS_NUM, out);
}
